// Frag2Mol_8100308320919
// MI455X (gfx1250) — matrix operand order repaired and checked by reading, not compiled, not run on hardware
//
#include <hip/hip_runtime.h>
#include <hip/hip_bf16.h>

// ---------------------------------------------------------------------------
// Frag2Mol forward for MI455X (gfx1250).
//  - heavy GEMMs: bf16 WMMA (v_wmma_f32_16x16x32_bf16), f32 accumulate,
//    64x256 block tile (16 waves, 2x2 WMMA tiles per wave), register
//    double-buffering, A staged in K-halves via the Tensor Data Mover
//    (2-D descriptor with LDS row padding against bank conflicts)
//  - logits stored with non-temporal hints (write-once 262 MB stream; keeps
//    the 33 MB bf16 W_out resident in the 192 MB L2)
//  - GRU scan: one persistent kernel per layer (8 co-resident workgroups,
//    per-step global arrival barrier)
// ---------------------------------------------------------------------------

typedef __attribute__((ext_vector_type(16))) __bf16 v16bf;
typedef __attribute__((ext_vector_type(8)))  __bf16 v8bf;
typedef __attribute__((ext_vector_type(8)))  float  v8f;
typedef __attribute__((ext_vector_type(4)))  unsigned int u32x4;
typedef __attribute__((ext_vector_type(8)))  int i32x8;
typedef __attribute__((ext_vector_type(4)))  int i32x4;

#define B_   32
#define S_   64
#define V_   32000
#define E_   128
#define H_   512
#define LAT_ 64
#define PAD_ 16            // bf16 elements of LDS row padding (= 8 dwords)

static __device__ __forceinline__ float sigm(float x) {
    return 1.0f / (1.0f + __expf(-x));
}

static __device__ __forceinline__ v8f wmma_bf16(v16bf a, v16bf b, v8f c) {
    return __builtin_amdgcn_wmma_f32_16x16x32_bf16(false, a, false, b,
                                                   (short)0, c, false, false);
}

union AFrag { v16bf v; v8bf q[2]; };

// Load one 16x32 A-fragment from an LDS row (row-major, padded stride).
static __device__ __forceinline__ void ldA(AFrag& f, const __bf16* row, int kk, int h) {
    f.q[0] = *(const v8bf*)(row + kk + 16 * h);       // K = kk+16h  .. +7
    f.q[1] = *(const v8bf*)(row + kk + 16 * h + 8);   // K = kk+16h+8.. +7
}

// ---------------------------------------------------------------------------
// Tensor Data Mover: 2-D tile of bf16 (width x rows, global row stride
// `stride` elements) -> LDS, padding 8 dwords after every 2<<icode dwords
// stored (i.e. after each row when 2<<icode == width/2).
// D# packing per CDNA5 ISA.
// ---------------------------------------------------------------------------
static __device__ __forceinline__ void tdm_load_2d(void* lds_dst, const void* gsrc,
                                                   int width, int rows, int stride,
                                                   int pad_icode) {
    u32x4 g0;
    unsigned long long ga = (unsigned long long)(uintptr_t)gsrc;
    g0[0] = 1u;                                           // count=1, user mode
    g0[1] = (unsigned int)(uintptr_t)lds_dst;             // lds_addr (low 32b)
    g0[2] = (unsigned int)ga;                             // global_addr[31:0]
    g0[3] = (unsigned int)((ga >> 32) & 0x01FFFFFFu) | (2u << 30);  // type=2

    unsigned int w0 = (1u << 16);                         // data_size = 2 bytes
    if (pad_icode >= 0)                                   // pad_enable + codes
        w0 |= (1u << 20) | ((unsigned)pad_icode << 22) | (7u << 25); // 8 dwords
    i32x8 g1;
    g1[0] = (int)w0;
    g1[1] = (int)(((unsigned)width & 0xFFFFu) << 16);                 // dim0 lo
    g1[2] = (int)(((unsigned)width >> 16) | (((unsigned)rows) << 16));// dim0 hi | dim1 lo
    g1[3] = (int)(((unsigned)width) << 16);               // dim1 hi=0 | tile_dim0
    g1[4] = rows;                                         // tile_dim1 | tile_dim2=0
    g1[5] = stride;                                       // dim0_stride[31:0]
    g1[6] = 0;
    g1[7] = 0;
    i32x4 z4 = {0, 0, 0, 0};
#if defined(__clang_major__) && (__clang_major__ >= 23)
    i32x8 z8 = {0, 0, 0, 0, 0, 0, 0, 0};
    __builtin_amdgcn_tensor_load_to_lds(g0, g1, z4, z4, z8, 0);
#else
    __builtin_amdgcn_tensor_load_to_lds(g0, g1, z4, z4, 0);
#endif
}

// ---------------------------------------------------------------------------
// f32 -> bf16 conversion (grid-stride)
// ---------------------------------------------------------------------------
__global__ void k_cvt_bf16(const float* __restrict__ s, __bf16* __restrict__ d,
                           size_t n) {
    size_t i  = (size_t)blockIdx.x * blockDim.x + threadIdx.x;
    size_t st = (size_t)gridDim.x * blockDim.x;
    for (; i < n; i += st) d[i] = (__bf16)s[i];
}

__global__ void k_zero_i32(int* __restrict__ p, int n) {
    int i = blockIdx.x * blockDim.x + threadIdx.x;
    if (i < n) p[i] = 0;
}

// ---------------------------------------------------------------------------
// Embedding gather + masked sum. grid=B, block=E(=128).
// ---------------------------------------------------------------------------
__global__ void k_embed(const int* __restrict__ inp, const float* __restrict__ tbl,
                        __bf16* __restrict__ embb, float* __restrict__ vf) {
    const int b = blockIdx.x, e = threadIdx.x;
    float acc = 0.0f;
    for (int t = 0; t < S_; ++t) {
        int tok = inp[b * S_ + t];
        float v = tbl[(size_t)tok * E_ + e];
        embb[((size_t)t * B_ + b) * E_ + e] = (__bf16)v;
        if (tok > 2) acc += v;
    }
    vf[b * E_ + e] = acc;
}

// ---------------------------------------------------------------------------
// mean / std / z
// ---------------------------------------------------------------------------
__global__ void k_enc(const float* __restrict__ vf,
                      const float* __restrict__ Wm, const float* __restrict__ bm,
                      const float* __restrict__ Wv, const float* __restrict__ bv,
                      const float* __restrict__ noise,
                      float* __restrict__ o_mean, float* __restrict__ o_std,
                      float* __restrict__ o_z) {
    int i = blockIdx.x * blockDim.x + threadIdx.x;
    if (i >= B_ * LAT_) return;
    int b = i / LAT_, la = i % LAT_;
    const float* x  = vf + b * E_;
    const float* wm = Wm + la * E_;
    const float* wv = Wv + la * E_;
    float m = bm[la], lv = bv[la];
    for (int k = 0; k < E_; ++k) { m += x[k] * wm[k]; lv += x[k] * wv[k]; }
    float sd = __expf(0.5f * lv);
    o_mean[i] = m;
    o_std[i]  = sd;
    for (int l = 0; l < 2; ++l)
        o_z[l * B_ * LAT_ + i] = noise[l * B_ * LAT_ + i] * sd + m;
}

// ---------------------------------------------------------------------------
// 3-layer MLP head (64->200->100->1). One block of 256 threads.
// ---------------------------------------------------------------------------
__global__ void k_mlp3(const float* __restrict__ mean,
                       const float* __restrict__ W1, const float* __restrict__ b1,
                       const float* __restrict__ W2, const float* __restrict__ b2,
                       const float* __restrict__ W3, const float* __restrict__ b3,
                       float* __restrict__ out) {
    __shared__ float h1[B_ * 200];
    __shared__ float h2[B_ * 100];
    const int tid = threadIdx.x;
    for (int i = tid; i < B_ * 200; i += 256) {
        int b = i / 200, j = i % 200;
        const float* x = mean + b * LAT_;
        const float* w = W1 + j * LAT_;
        float a = b1[j];
        for (int k = 0; k < LAT_; ++k) a += x[k] * w[k];
        h1[i] = fmaxf(a, 0.0f);
    }
    __syncthreads();
    for (int i = tid; i < B_ * 100; i += 256) {
        int b = i / 100, j = i % 100;
        const float* x = h1 + b * 200;
        const float* w = W2 + j * 200;
        float a = b2[j];
        for (int k = 0; k < 200; ++k) a += x[k] * w[k];
        h2[i] = fmaxf(a, 0.0f);
    }
    __syncthreads();
    for (int i = tid; i < B_; i += 256) {
        const float* x = h2 + i * 100;
        float a = b3[0];
        for (int k = 0; k < 100; ++k) a += x[k] * W3[k];
        out[i] = a;
    }
}

// ---------------------------------------------------------------------------
// state = z @ W_l2r.T + b  -> initial GRU hidden states (f32 + bf16)
// ---------------------------------------------------------------------------
__global__ void k_state(const float* __restrict__ z, const float* __restrict__ W,
                        const float* __restrict__ bias,
                        float* __restrict__ h0f, __bf16* __restrict__ h0b) {
    int i = blockIdx.x * blockDim.x + threadIdx.x;
    if (i >= 2 * B_ * H_) return;
    int l = i / (B_ * H_), rem = i % (B_ * H_);
    int b = rem / H_, j = rem % H_;
    const float* zz = z + (l * B_ + b) * LAT_;
    const float* w  = W + j * LAT_;
    float a = bias[j];
    for (int k = 0; k < LAT_; ++k) a += zz[k] * w[k];
    h0f[i] = a;
    h0b[i] = (__bf16)a;
}

// ---------------------------------------------------------------------------
// Generic bf16 GEMM: C[m,n] = sum_k A[m,k]*W[n,k] (+bias[n]).
// A bf16 [M,K] row-major, W bf16 [N,K] row-major, C f32 [*,N].
// Block: 512 threads (16 waves); tile 64(M) x 256(N); wave = 32x32 (2x2 WMMA
// tiles). A staged per K-half (<=256) into padded LDS by TDM 2-D descriptor.
// Requires M%64==0, N%256==0, K%64==0 and (K<=256 or K%256==0).
// remap!=0: om = (m%32)*S_ + m/32 (t*B+b -> b*S+t) and non-temporal stores.
// ---------------------------------------------------------------------------
__global__ void k_gemm_bf16(const __bf16* __restrict__ A,
                            const __bf16* __restrict__ W,
                            float* __restrict__ C, int K, int N,
                            const float* __restrict__ bias, int remap) {
    extern __shared__ __bf16 sA[];               // [64][KS + PAD_]
    const int tid    = threadIdx.x;
    const int m_base = blockIdx.y * 64;
    const int n_base = blockIdx.x * 256;
    const int KS     = (K < 256) ? K : 256;      // staged K per pass
    const int KSP    = KS + PAD_;
    const int wave   = tid >> 5, lane = tid & 31;
    const int h = lane >> 4, l16 = lane & 15;
    const int mgrp = wave >> 3;                  // 0/1 -> rows +0 / +32
    const int ns   = wave & 7;                   // 8 N-slabs of 32

    const int icode = 31 - __clz(KS) - 2;        // 2<<icode == KS/2 dwords/row
    const int n0 = n_base + ns * 32;
    const __bf16* a0p = sA + (size_t)(mgrp * 32 + l16) * KSP;
    const __bf16* a1p = sA + (size_t)(mgrp * 32 + 16 + l16) * KSP;
    const __bf16* w0p = W + (size_t)(n0 + l16) * K + 16 * h;
    const __bf16* w1p = W + (size_t)(n0 + 16 + l16) * K + 16 * h;

    v8f acc00, acc01, acc10, acc11;
#pragma unroll
    for (int r = 0; r < 8; ++r) { acc00[r] = 0.f; acc01[r] = 0.f; acc10[r] = 0.f; acc11[r] = 0.f; }

    for (int kb = 0; kb < K; kb += KS) {
        if (wave == 0) {                         // TDM: 64 x KS tile, padded
            tdm_load_2d(sA, A + (size_t)m_base * K + kb, KS, 64, K, icode);
            __builtin_amdgcn_s_wait_tensorcnt(0);
        }
        __syncthreads();

        AFrag a0c, a1c, a0n, a1n;
        v16bf b0c, b1c, b0n, b1n;
        ldA(a0c, a0p, 0, h); ldA(a1c, a1p, 0, h);
        b0c = *(const v16bf*)(w0p + kb);
        b1c = *(const v16bf*)(w1p + kb);

        for (int kk = 0; kk < KS; kk += 64) {
            const int k1 = kk + 32;
            ldA(a0n, a0p, k1, h); ldA(a1n, a1p, k1, h);
            b0n = *(const v16bf*)(w0p + kb + k1);
            b1n = *(const v16bf*)(w1p + kb + k1);
            acc00 = wmma_bf16(a0c.v, b0c, acc00);
            acc01 = wmma_bf16(a0c.v, b1c, acc01);
            acc10 = wmma_bf16(a1c.v, b0c, acc10);
            acc11 = wmma_bf16(a1c.v, b1c, acc11);

            const int k2 = (k1 + 32 < KS) ? (k1 + 32) : 0;  // clamped prefetch
            ldA(a0c, a0p, k2, h); ldA(a1c, a1p, k2, h);
            b0c = *(const v16bf*)(w0p + kb + k2);
            b1c = *(const v16bf*)(w1p + kb + k2);
            acc00 = wmma_bf16(a0n.v, b0n, acc00);
            acc01 = wmma_bf16(a0n.v, b1n, acc01);
            acc10 = wmma_bf16(a1n.v, b0n, acc10);
            acc11 = wmma_bf16(a1n.v, b1n, acc11);
        }
        __syncthreads();                         // protect sA before next TDM
    }

    const int nA = n0 + l16, nB = n0 + 16 + l16;
    const float bbA = bias ? bias[nA] : 0.0f;
    const float bbB = bias ? bias[nB] : 0.0f;
#pragma unroll
    for (int r = 0; r < 8; ++r) {
        int m0 = m_base + mgrp * 32 + 8 * h + r;   // mi = 0
        int m1 = m0 + 16;                          // mi = 1
        if (remap) {
            int o0 = (m0 & (B_ - 1)) * S_ + (m0 >> 5);
            int o1 = (m1 & (B_ - 1)) * S_ + (m1 >> 5);
            __builtin_nontemporal_store(acc00[r] + bbA, &C[(size_t)o0 * N + nA]);
            __builtin_nontemporal_store(acc01[r] + bbB, &C[(size_t)o0 * N + nB]);
            __builtin_nontemporal_store(acc10[r] + bbA, &C[(size_t)o1 * N + nA]);
            __builtin_nontemporal_store(acc11[r] + bbB, &C[(size_t)o1 * N + nB]);
        } else {
            C[(size_t)m0 * N + nA] = acc00[r] + bbA;
            C[(size_t)m0 * N + nB] = acc01[r] + bbB;
            C[(size_t)m1 * N + nA] = acc10[r] + bbA;
            C[(size_t)m1 * N + nB] = acc11[r] + bbB;
        }
    }
}

// ---------------------------------------------------------------------------
// Persistent GRU scan for one layer: 8 co-resident workgroups, each owning
// 64 hidden columns; 64 time steps with a global arrival barrier per step.
// h double-buffered by step parity. hW = h @ Whh.T via WMMA (waves 0..5).
// ---------------------------------------------------------------------------
__global__ void k_gru_scan(const float* __restrict__ xW,     // [S*B][3H]
                           const float* __restrict__ h0f, const __bf16* __restrict__ h0b,
                           const __bf16* __restrict__ Whh,   // [3H][H] bf16
                           const float* __restrict__ bih, const float* __restrict__ bhh,
                           const int* __restrict__ lengths,
                           float* __restrict__ hf0, float* __restrict__ hf1,
                           __bf16* __restrict__ hb0, __bf16* __restrict__ hb1,
                           __bf16* __restrict__ outb,        // [S*B][H]
                           int* __restrict__ barcnt) {       // [S_], pre-zeroed
    constexpr int HP = H_ + PAD_;
    __shared__ __bf16 sA[B_ * HP];          // padded h slab (bf16)
    __shared__ float  sHW[3][B_][64];

    const int tid  = threadIdx.x;
    const int c0   = blockIdx.x * 64;
    const int nwg  = gridDim.x;
    const int wave = tid >> 5, lane = tid & 31;
    const int h = lane >> 4, l16 = lane & 15;

    for (int t = 0; t < S_; ++t) {
        const __bf16* hpb = (t == 0) ? h0b : (((t - 1) & 1) ? hb1 : hb0);
        const float*  hpf = (t == 0) ? h0f : (((t - 1) & 1) ? hf1 : hf0);
        float*  hnf = (t & 1) ? hf1 : hf0;
        __bf16* hnb = (t & 1) ? hb1 : hb0;

        if (wave == 0) {                     // TDM: h slab -> padded LDS
            tdm_load_2d(sA, hpb, H_, B_, H_, 7);   // 256 dwords per row
            __builtin_amdgcn_s_wait_tensorcnt(0);
        }
        __syncthreads();

        if (wave < 6) {
            const int g = wave >> 1, mt = wave & 1;
            const __bf16* arow = sA + (size_t)(mt * 16 + l16) * HP;
            const __bf16* wp[4];
#pragma unroll
            for (int j = 0; j < 4; ++j)
                wp[j] = Whh + (size_t)(g * H_ + c0 + j * 16 + l16) * H_ + 16 * h;

            v8f acc[4];
#pragma unroll
            for (int j = 0; j < 4; ++j)
#pragma unroll
                for (int r = 0; r < 8; ++r) acc[j][r] = 0.0f;

            AFrag ac, an;
            v16bf bc[4], bn[4];
            ldA(ac, arow, 0, h);
#pragma unroll
            for (int j = 0; j < 4; ++j) bc[j] = *(const v16bf*)(wp[j]);

            for (int kk = 0; kk < H_; kk += 64) {
                const int k1 = kk + 32;
                ldA(an, arow, k1, h);
#pragma unroll
                for (int j = 0; j < 4; ++j) bn[j] = *(const v16bf*)(wp[j] + k1);
#pragma unroll
                for (int j = 0; j < 4; ++j) acc[j] = wmma_bf16(ac.v, bc[j], acc[j]);

                const int k2 = (k1 + 32 < H_) ? (k1 + 32) : 0;
                ldA(ac, arow, k2, h);
#pragma unroll
                for (int j = 0; j < 4; ++j) bc[j] = *(const v16bf*)(wp[j] + k2);
#pragma unroll
                for (int j = 0; j < 4; ++j) acc[j] = wmma_bf16(an.v, bn[j], acc[j]);
            }
#pragma unroll
            for (int j = 0; j < 4; ++j)
#pragma unroll
                for (int r = 0; r < 8; ++r)
                    sHW[g][mt * 16 + 8 * h + r][j * 16 + l16] = acc[j][r];
        }
        __syncthreads();

        for (int i = tid; i < B_ * 64; i += 256) {
            int b = i >> 6, j = i & 63, col = c0 + j;
            float hr = sHW[0][b][j] + bhh[col];
            float hz = sHW[1][b][j] + bhh[H_ + col];
            float hn = sHW[2][b][j] + bhh[2 * H_ + col];
            const float* xrow = xW + ((size_t)t * B_ + b) * (3 * H_);
            float xr = xrow[col]          + bih[col];
            float xz = xrow[H_ + col]     + bih[H_ + col];
            float xn = xrow[2 * H_ + col] + bih[2 * H_ + col];
            float r = sigm(xr + hr);
            float u = sigm(xz + hz);
            float n = tanhf(xn + r * hn);
            float hold = hpf[b * H_ + col];
            float hnew = (1.0f - u) * n + u * hold;
            bool  msk  = t < lengths[b];
            float hsel = msk ? hnew : hold;
            hnf[b * H_ + col] = hsel;
            hnb[b * H_ + col] = (__bf16)hsel;
            outb[((size_t)t * B_ + b) * H_ + col] = (__bf16)(msk ? hnew : 0.0f);
        }

        // global step barrier (8 workgroups, guaranteed co-resident)
        __threadfence();
        __syncthreads();
        if (tid == 0) {
            __hip_atomic_fetch_add(&barcnt[t], 1, __ATOMIC_ACQ_REL,
                                   __HIP_MEMORY_SCOPE_AGENT);
            while (__hip_atomic_load(&barcnt[t], __ATOMIC_ACQUIRE,
                                     __HIP_MEMORY_SCOPE_AGENT) < nwg)
                __builtin_amdgcn_s_sleep(1);
        }
        __syncthreads();
        __threadfence();
    }
}

// ---------------------------------------------------------------------------
// Host side
// ---------------------------------------------------------------------------
extern "C" void kernel_launch(void* const* d_in, const int* in_sizes, int n_in,
                              void* d_out, int out_size, void* d_ws, size_t ws_size,
                              hipStream_t stream) {
    (void)in_sizes; (void)n_in; (void)out_size; (void)ws_size;

    const int*   inputs  = (const int*)d_in[0];
    const int*   lengths = (const int*)d_in[1];
    const float* noise   = (const float*)d_in[2];
    const float* etab    = (const float*)d_in[3];
    const float* enc_Wm  = (const float*)d_in[4];
    const float* enc_bm  = (const float*)d_in[5];
    const float* enc_Wv  = (const float*)d_in[6];
    const float* enc_bv  = (const float*)d_in[7];
    const float* W_l2r   = (const float*)d_in[8];
    const float* b_l2r   = (const float*)d_in[9];
    const float* Wih0    = (const float*)d_in[10];
    const float* Whh0    = (const float*)d_in[11];
    const float* bih0    = (const float*)d_in[12];
    const float* bhh0    = (const float*)d_in[13];
    const float* Wih1    = (const float*)d_in[14];
    const float* Whh1    = (const float*)d_in[15];
    const float* bih1    = (const float*)d_in[16];
    const float* bhh1    = (const float*)d_in[17];
    const float* W_out   = (const float*)d_in[18];
    const float* b_out   = (const float*)d_in[19];
    const float* lp_W1   = (const float*)d_in[20];
    const float* lp_b1   = (const float*)d_in[21];
    const float* lp_W2   = (const float*)d_in[22];
    const float* lp_b2   = (const float*)d_in[23];
    const float* lp_W3   = (const float*)d_in[24];
    const float* lp_b3   = (const float*)d_in[25];
    const float* sa_W1   = (const float*)d_in[26];
    const float* sa_b1   = (const float*)d_in[27];
    const float* sa_W2   = (const float*)d_in[28];
    const float* sa_b2   = (const float*)d_in[29];
    const float* sa_W3   = (const float*)d_in[30];
    const float* sa_b3   = (const float*)d_in[31];

    float* o_output = (float*)d_out;
    float* o_mean   = o_output + (size_t)B_ * S_ * V_;
    float* o_std    = o_mean + B_ * LAT_;
    float* o_z      = o_std + B_ * LAT_;
    float* o_logp   = o_z + 2 * B_ * LAT_;
    float* o_sas    = o_logp + B_;

    char* wp = (char*)d_ws;
    auto alloc = [&](size_t bytes) -> void* {
        void* r = wp;
        wp += (bytes + 255) & ~(size_t)255;
        return r;
    };
    __bf16* embb  = (__bf16*)alloc((size_t)S_ * B_ * E_ * 2);
    float*  vf    = (float*) alloc((size_t)B_ * E_ * 4);
    __bf16* Wih0b = (__bf16*)alloc((size_t)3 * H_ * E_ * 2);
    __bf16* Whh0b = (__bf16*)alloc((size_t)3 * H_ * H_ * 2);
    __bf16* Wih1b = (__bf16*)alloc((size_t)3 * H_ * H_ * 2);
    __bf16* Whh1b = (__bf16*)alloc((size_t)3 * H_ * H_ * 2);
    __bf16* Woutb = (__bf16*)alloc((size_t)V_ * H_ * 2);
    float*  xW0   = (float*) alloc((size_t)S_ * B_ * 3 * H_ * 4);
    float*  xW1   = (float*) alloc((size_t)S_ * B_ * 3 * H_ * 4);
    __bf16* out0b = (__bf16*)alloc((size_t)S_ * B_ * H_ * 2);
    __bf16* out1b = (__bf16*)alloc((size_t)S_ * B_ * H_ * 2);
    float*  h0f   = (float*) alloc((size_t)2 * B_ * H_ * 4);
    __bf16* h0b   = (__bf16*)alloc((size_t)2 * B_ * H_ * 2);
    float*  hf0   = (float*) alloc((size_t)B_ * H_ * 4);
    float*  hf1   = (float*) alloc((size_t)B_ * H_ * 4);
    __bf16* hb0   = (__bf16*)alloc((size_t)B_ * H_ * 2);
    __bf16* hb1   = (__bf16*)alloc((size_t)B_ * H_ * 2);
    int*    bar0  = (int*)   alloc((size_t)S_ * 4);
    int*    bar1  = (int*)   alloc((size_t)S_ * 4);

    // barrier counters must be zero at the start of every call
    k_zero_i32<<<1, 128, 0, stream>>>(bar0, S_);
    k_zero_i32<<<1, 128, 0, stream>>>(bar1, S_);

    // weight conversions to bf16
    k_cvt_bf16<<<512, 256, 0, stream>>>(Wih0, Wih0b, (size_t)3 * H_ * E_);
    k_cvt_bf16<<<1024, 256, 0, stream>>>(Whh0, Whh0b, (size_t)3 * H_ * H_);
    k_cvt_bf16<<<1024, 256, 0, stream>>>(Wih1, Wih1b, (size_t)3 * H_ * H_);
    k_cvt_bf16<<<1024, 256, 0, stream>>>(Whh1, Whh1b, (size_t)3 * H_ * H_);
    k_cvt_bf16<<<4096, 256, 0, stream>>>(W_out, Woutb, (size_t)V_ * H_);

    // embedding + masked sum
    k_embed<<<B_, E_, 0, stream>>>(inputs, etab, embb, vf);

    // encoder mean/std/z (straight into d_out slices)
    k_enc<<<(B_ * LAT_ + 255) / 256, 256, 0, stream>>>(
        vf, enc_Wm, enc_bm, enc_Wv, enc_bv, noise, o_mean, o_std, o_z);

    // property heads
    k_mlp3<<<1, 256, 0, stream>>>(o_mean, lp_W1, lp_b1, lp_W2, lp_b2, lp_W3, lp_b3, o_logp);
    k_mlp3<<<1, 256, 0, stream>>>(o_mean, sa_W1, sa_b1, sa_W2, sa_b2, sa_W3, sa_b3, o_sas);

    // initial hidden states
    k_state<<<(2 * B_ * H_ + 255) / 256, 256, 0, stream>>>(o_z, W_l2r, b_l2r, h0f, h0b);

    // xW0 = emb @ Wih0.T (M=2048, K=128, N=1536); KS=128
    k_gemm_bf16<<<dim3(3 * H_ / 256, S_ * B_ / 64), 512,
                  (size_t)64 * (E_ + PAD_) * 2, stream>>>(
        embb, Wih0b, xW0, E_, 3 * H_, nullptr, 0);

    // GRU layer 0 (persistent scan, 8 workgroups)
    k_gru_scan<<<H_ / 64, 256, 0, stream>>>(
        xW0, h0f, h0b, Whh0b, bih0, bhh0, lengths,
        hf0, hf1, hb0, hb1, out0b, bar0);

    // xW1 = out0 @ Wih1.T (M=2048, K=512, N=1536); KS=256
    k_gemm_bf16<<<dim3(3 * H_ / 256, S_ * B_ / 64), 512,
                  (size_t)64 * (256 + PAD_) * 2, stream>>>(
        out0b, Wih1b, xW1, H_, 3 * H_, nullptr, 0);

    // GRU layer 1
    k_gru_scan<<<H_ / 64, 256, 0, stream>>>(
        xW1, h0f + (size_t)B_ * H_, h0b + (size_t)B_ * H_, Whh1b, bih1, bhh1,
        lengths, hf0, hf1, hb0, hb1, out1b, bar1);

    // logits = out1 @ W_out.T + b_out (M=2048, K=512, N=32000); KS=256, NT
    k_gemm_bf16<<<dim3(V_ / 256, S_ * B_ / 64), 512,
                  (size_t)64 * (256 + PAD_) * 2, stream>>>(
        out1b, Woutb, o_output, H_, V_, b_out, 1);
}
